// TT_76733885710854
// MI455X (gfx1250) — matrix operand order repaired and checked by reading, not compiled, not run on hardware
//
#include <hip/hip_runtime.h>
#include <hip/hip_bf16.h>

#define IN_DIM   4096
#define OUT_DIM  4096
#define BS       8192            // 4 * 2048
#define KT_STEPS (IN_DIM / 32)   // 128 k-tiles of 32

typedef __bf16 bf16x16 __attribute__((ext_vector_type(16)));
typedef float  v8f     __attribute__((ext_vector_type(8)));

// Native RNE conversions (lower to v_cvt_* on gfx1250)
static __device__ __forceinline__ __bf16 to_bf16(float f)  { return (__bf16)f; }
static __device__ __forceinline__ float  to_f32(__bf16 b)  { return (float)b; }

// =====================================================================
// Kernel 0: pre-convert x into bf16 hi/lo, stored pre-swizzled in the
// WMMA 16-bit A-fragment layout (16x32 tile per (mt,kt)):
//   lane l: row = mt*16 + (l&15), half h = l>>4
//   vector elem e (=2j+p): K = kt*32 + ((e>>3)&1)*16 + h*8 + ((e>>1)&3)*2 + (e&1)
//   flat = ((mt*KT_STEPS + kt)*32 + lane)*16 + e
// =====================================================================
__global__ __launch_bounds__(256) void tt_conv_x(
    const float* __restrict__ x,
    __bf16* __restrict__ xhi, __bf16* __restrict__ xlo)
{
    const int gid  = blockIdx.x * 256 + threadIdx.x;
    const int lane = gid & 31;
    const int frag = gid >> 5;                 // mt*KT_STEPS + kt
    const int kt   = frag & (KT_STEPS - 1);
    const int mt   = frag / KT_STEPS;
    const int row  = mt * 16 + (lane & 15);
    const int h    = lane >> 4;
    const float* xrow = x + (long)row * IN_DIM;

    bf16x16 a_hi, a_lo;
#pragma unroll
    for (int j = 0; j < 8; ++j) {
        const int k = kt * 32 + ((j >> 2) << 4) + h * 8 + ((j & 3) << 1);
        const float2 xv = *(const float2*)(xrow + k);
        const __bf16 h0 = to_bf16(xv.x);
        const __bf16 h1 = to_bf16(xv.y);
        a_hi[2 * j]     = h0;
        a_hi[2 * j + 1] = h1;
        a_lo[2 * j]     = to_bf16(xv.x - to_f32(h0));
        a_lo[2 * j + 1] = to_bf16(xv.y - to_f32(h1));
    }
    const long base = (long)gid * 16;
    *(bf16x16*)(xhi + base) = a_hi;
    *(bf16x16*)(xlo + base) = a_lo;
}

// =====================================================================
// Kernel 1: materialize G from TT cores, split into bf16 hi/lo, and
// store pre-swizzled in the WMMA B-fragment layout:
//   fragment(nt, kt): lane = ((kin>>3)&1)*16 + nin, elem e = (kin>>4)*8 + (kin & 7)
//   flat = ((nt*KT_STEPS + kt)*32 + lane)*16 + e
// One block per (i,j); r = ((i*16+j)*16)+k, q = ((n*16+m)*16)+o.
// G[r][q] = sum_{b,c} core0[i,n,b] * core1[b,j,m,c] * core2[c,k,o]
// =====================================================================
__global__ __launch_bounds__(256) void tt_build_g(
    const float* __restrict__ core0,   // [1,16,16,16]  (i,n,b)
    const float* __restrict__ core1,   // [16,16,16,16] (b,j,m,c)
    const float* __restrict__ core2,   // [16,16,16,1]  (c,k,o)
    __bf16* __restrict__ ghi, __bf16* __restrict__ glo)
{
    __shared__ float sC0[256];    // core0[i][n][b]           -> [n*16+b]
    __shared__ float sC1[4096];   // core1[b][j][m][c]        -> [b*256+m*16+c]
    __shared__ float sC2[4096];   // core2[c][k][o]           -> [c*256+k*16+o]
    __shared__ float sU [4096];   // U[n][m][c] = sum_b c0*c1 -> [n*256+m*16+c]

    const int tid = threadIdx.x;
    const int bi  = blockIdx.x >> 4;   // i
    const int bj  = blockIdx.x & 15;   // j

    if (tid < 256) sC0[tid] = core0[(bi * 16 + (tid >> 4)) * 16 + (tid & 15)];
    for (int idx = tid; idx < 4096; idx += 256) {
        sC2[idx] = core2[idx];
        int b = idx >> 8, m = (idx >> 4) & 15, c = idx & 15;
        sC1[idx] = core1[((b * 16 + bj) * 16 + m) * 16 + c];
    }
    __syncthreads();

    for (int idx = tid; idx < 4096; idx += 256) {
        int n = idx >> 8, m = (idx >> 4) & 15, c = idx & 15;
        float acc = 0.f;
#pragma unroll
        for (int b = 0; b < 16; ++b)
            acc += sC0[n * 16 + b] * sC1[b * 256 + m * 16 + c];
        sU[idx] = acc;
    }
    __syncthreads();

    const int n = tid >> 4;
    const int k = tid & 15;
    const int r  = (bi * 16 + bj) * 16 + k;
    const int nt = r >> 4;
    const int nin = r & 15;

    for (int m = 0; m < 16; ++m) {
        for (int o = 0; o < 16; ++o) {
            float acc = 0.f;
#pragma unroll
            for (int c = 0; c < 16; ++c)
                acc += sU[n * 256 + m * 16 + c] * sC2[c * 256 + k * 16 + o];
            const int q   = (n * 16 + m) * 16 + o;
            const int kt  = q >> 5;
            const int kin = q & 31;
            const int lane = (((kin >> 3) & 1) << 4) | nin;
            const int e    = ((kin >> 4) << 3) | (kin & 7);
            const long idx = ((((long)nt * KT_STEPS + kt) * 32) + lane) * 16 + e;
            const __bf16 hi = to_bf16(acc);
            ghi[idx] = hi;
            glo[idx] = to_bf16(acc - to_f32(hi));
        }
    }
}

// =====================================================================
// Kernel 2: out[BS,4096] = x @ G^T + bias via bf16 hi/lo WMMA.
// Block: 256 thr = 8 wave32 arranged 4M x 2N  -> block tile 64 x 256.
// Wave tile: 16(M) x 128(N) = 8 x v8f accumulators; K-step 32;
// 24 v_wmma per K-step (3 products per subtile).
// PRECONV: A fragments read pre-swizzled from workspace (zero VALU in loop);
// else: built from fp32 x with native cvt.
// =====================================================================
template <bool PRECONV>
__global__ __launch_bounds__(256) void tt_gemm(
    const float*  __restrict__ x,
    const __bf16* __restrict__ xhi,
    const __bf16* __restrict__ xlo,
    const __bf16* __restrict__ ghi,
    const __bf16* __restrict__ glo,
    const float*  __restrict__ bias,
    float* __restrict__ out)
{
    const int lane = threadIdx.x & 31;
    const int w    = threadIdx.x >> 5;
    const int wm   = w >> 1;                       // 0..3
    const int wn   = w & 1;                        // 0..1
    const int m0   = blockIdx.y * 64 + wm * 16;
    const int n0   = blockIdx.x * 256 + wn * 128;
    const int h    = lane >> 4;
    const int mt   = m0 >> 4;                      // A fragment row-tile
    const int ntb  = n0 >> 4;                      // first B n-tile
    const float* xrow = x + (long)(m0 + (lane & 15)) * IN_DIM;

    v8f acc[8] = {};

    for (int kt = 0; kt < KT_STEPS; ++kt) {
        bf16x16 a_hi, a_lo;
        if (PRECONV) {
            const long abase = (((long)mt * KT_STEPS + kt) * 32 + lane) * 16;
            a_hi = *(const bf16x16*)(xhi + abase);
            a_lo = *(const bf16x16*)(xlo + abase);
        } else {
#pragma unroll
            for (int j = 0; j < 8; ++j) {
                const int k = kt * 32 + ((j >> 2) << 4) + h * 8 + ((j & 3) << 1);
                const float2 xv = *(const float2*)(xrow + k);
                const __bf16 h0 = to_bf16(xv.x);
                const __bf16 h1 = to_bf16(xv.y);
                a_hi[2 * j]     = h0;
                a_hi[2 * j + 1] = h1;
                a_lo[2 * j]     = to_bf16(xv.x - to_f32(h0));
                a_lo[2 * j + 1] = to_bf16(xv.y - to_f32(h1));
            }
        }
#pragma unroll
        for (int s = 0; s < 8; ++s) {
            const long bbase = (((long)(ntb + s) * KT_STEPS + kt) * 32 + lane) * 16;
            const bf16x16 b_hi = *(const bf16x16*)(ghi + bbase);
            const bf16x16 b_lo = *(const bf16x16*)(glo + bbase);
            acc[s] = __builtin_amdgcn_wmma_f32_16x16x32_bf16(
                false, a_hi, false, b_hi, (short)0, acc[s], false, false);
            acc[s] = __builtin_amdgcn_wmma_f32_16x16x32_bf16(
                false, a_lo, false, b_hi, (short)0, acc[s], false, false);
            acc[s] = __builtin_amdgcn_wmma_f32_16x16x32_bf16(
                false, a_hi, false, b_lo, (short)0, acc[s], false, false);
        }
    }

    // C/D layout: lane half h -> M = m0 + h*8 + v, N = n0 + s*16 + (lane&15)
#pragma unroll
    for (int s = 0; s < 8; ++s) {
        const int col = n0 + s * 16 + (lane & 15);
        const float bv = bias[col];
#pragma unroll
        for (int v = 0; v < 8; ++v) {
            const int rowOut = m0 + h * 8 + v;
            out[(long)rowOut * OUT_DIM + col] = acc[s][v] + bv;
        }
    }
}

extern "C" void kernel_launch(void* const* d_in, const int* in_sizes, int n_in,
                              void* d_out, int out_size, void* d_ws, size_t ws_size,
                              hipStream_t stream) {
    const float* x     = (const float*)d_in[0];
    const float* core0 = (const float*)d_in[1];
    const float* core1 = (const float*)d_in[2];
    const float* core2 = (const float*)d_in[3];
    const float* bias  = (const float*)d_in[4];
    float* out = (float*)d_out;
    (void)in_sizes; (void)n_in; (void)out_size;

    const size_t gElems = (size_t)OUT_DIM * IN_DIM;   // 16M
    const size_t xElems = (size_t)BS * IN_DIM;        // 33.5M
    __bf16* ghi = (__bf16*)d_ws;                      // 32 MB
    __bf16* glo = ghi + gElems;                       // 32 MB
    __bf16* xhi = glo + gElems;                       // 64 MB
    __bf16* xlo = xhi + xElems;                       // 64 MB
    const bool preconv = ws_size >= 2 * sizeof(__bf16) * (gElems + xElems); // 192 MB

    // Build swizzled bf16 hi/lo G (256 blocks = one per (i,j))
    tt_build_g<<<dim3(256), dim3(256), 0, stream>>>(core0, core1, core2, ghi, glo);

    if (preconv) {
        // Pre-swizzle x into A fragments: (BS/16)*KT_STEPS frags * 32 lanes threads
        tt_conv_x<<<dim3((BS / 16) * KT_STEPS * 32 / 256), dim3(256), 0, stream>>>(x, xhi, xlo);
        tt_gemm<true><<<dim3(OUT_DIM / 256, BS / 64), dim3(256), 0, stream>>>(
            x, xhi, xlo, ghi, glo, bias, out);
    } else {
        tt_gemm<false><<<dim3(OUT_DIM / 256, BS / 64), dim3(256), 0, stream>>>(
            x, xhi, xlo, ghi, glo, bias, out);
    }
}
